// PAM_Module_75359496175775
// MI455X (gfx1250) — compile-verified, matrix operand order checked by reading, not run on hardware
//
#include <hip/hip_runtime.h>

typedef unsigned short u16;
typedef __attribute__((ext_vector_type(16))) __bf16 v16bf;
typedef __attribute__((ext_vector_type(8)))  __bf16 v8bf;
typedef __attribute__((ext_vector_type(8)))  float  v8f;

#define BATCH 4
#define CH    512
#define C8    64
#define HH    64
#define WW    64
#define NPOS  4096   // H*W
#define NT    256    // NPOS/16

static __device__ __forceinline__ u16 f2bf(float f) {
  union { float f; unsigned int u; } v; v.f = f;
  unsigned int u = v.u;
  u += 0x7FFFu + ((u >> 16) & 1u);   // round-to-nearest-even
  return (u16)(u >> 16);
}

// Load a 16-element bf16 WMMA fragment (A-style row / B-style column) from a
// contiguous-K row pointer.  Lane halves follow the CDNA5 16-bit A/B layout:
// lanes 0-15 hold K = kb+{0..7, 16..23}, lanes 16-31 hold K = kb+{8..15, 24..31}.
static __device__ __forceinline__ v16bf ldfrag(const u16* rowp, int kb, int half) {
  const v8bf lo = *reinterpret_cast<const v8bf*>(rowp + kb + half * 8);
  const v8bf hi = *reinterpret_cast<const v8bf*>(rowp + kb + 16 + half * 8);
  v16bf r;
#pragma unroll
  for (int i = 0; i < 8; ++i) { r[i] = lo[i]; r[8 + i] = hi[i]; }
  return r;
}

#define WMMA_BF16(a, b, c) \
  __builtin_amdgcn_wmma_f32_16x16x32_bf16(false, (a), false, (b), (short)0, (c), false, false)

// ---------------------------------------------------------------- prep kernels
__global__ void xpose_kernel(const float* __restrict__ x, u16* __restrict__ xT) {
  __shared__ float tile[32][33];
  int b = blockIdx.z;
  int n0 = blockIdx.x * 32, c0 = blockIdx.y * 32;
  for (int i = threadIdx.y; i < 32; i += 8)
    tile[i][threadIdx.x] = x[((size_t)(b * CH + c0 + i)) * NPOS + n0 + threadIdx.x];
  __syncthreads();
  for (int i = threadIdx.y; i < 32; i += 8)
    xT[((size_t)(b * NPOS + n0 + i)) * CH + c0 + threadIdx.x] = f2bf(tile[threadIdx.x][i]);
}

__global__ void cvt_kernel(const float* __restrict__ s, u16* __restrict__ d, int n) {
  int i = blockIdx.x * blockDim.x + threadIdx.x;
  if (i < n) d[i] = f2bf(s[i]);
}

// ------------------------------------------------------- conv1x1 WMMA GEMMs
// out[b][row][col] = sum_c W[row][c] * x[b][c][col] + bias[row]
__global__ void gemm_conv_f32(const u16* __restrict__ Wb, const u16* __restrict__ xT,
                              const float* __restrict__ bias, float* __restrict__ out, int M) {
  int b = blockIdx.y;
  int w = threadIdx.x >> 5, l = threadIdx.x & 31;
  int lh = l >> 4, ll = l & 15;
  int t = blockIdx.x * 8 + w;
  int mt = t / NT, nt = t % NT;
  const u16* arow = Wb + (size_t)(mt * 16 + ll) * CH;
  const u16* brow = xT + ((size_t)b * NPOS + nt * 16 + ll) * CH;
  v8f acc = {};
#pragma unroll 4
  for (int kb = 0; kb < CH; kb += 32) {
    v16bf a  = ldfrag(arow, kb, lh);
    v16bf bb = ldfrag(brow, kb, lh);
    acc = WMMA_BF16(a, bb, acc);
  }
  int col = nt * 16 + ll;
#pragma unroll
  for (int i = 0; i < 8; ++i) {
    int row = mt * 16 + 8 * lh + i;
    out[((size_t)(b * M + row)) * NPOS + col] = acc[i] + bias[row];
  }
}

__global__ void gemm_conv_bf16(const u16* __restrict__ Wb, const u16* __restrict__ xT,
                               const float* __restrict__ bias, u16* __restrict__ out, int M) {
  int b = blockIdx.y;
  int w = threadIdx.x >> 5, l = threadIdx.x & 31;
  int lh = l >> 4, ll = l & 15;
  int t = blockIdx.x * 8 + w;
  int mt = t / NT, nt = t % NT;
  const u16* arow = Wb + (size_t)(mt * 16 + ll) * CH;
  const u16* brow = xT + ((size_t)b * NPOS + nt * 16 + ll) * CH;
  v8f acc = {};
#pragma unroll 4
  for (int kb = 0; kb < CH; kb += 32) {
    v16bf a  = ldfrag(arow, kb, lh);
    v16bf bb = ldfrag(brow, kb, lh);
    acc = WMMA_BF16(a, bb, acc);
  }
  int col = nt * 16 + ll;
#pragma unroll
  for (int i = 0; i < 8; ++i) {
    int row = mt * 16 + 8 * lh + i;
    out[((size_t)(b * M + row)) * NPOS + col] = f2bf(acc[i] + bias[row]);
  }
}

// ----------------------------------------------- 3x3 maxpool + transpose->bf16
__global__ void pool_t_kernel(const float* __restrict__ qc, u16* __restrict__ qpT) {
  int tid = blockIdx.x * blockDim.x + threadIdx.x;
  int n  = tid & (NPOS - 1);
  int ch = (tid >> 12) & (C8 - 1);
  int b  = tid >> 18;
  int h = n >> 6, w = n & 63;
  const float* base = qc + ((size_t)(b * C8 + ch)) * NPOS;
  float m = -1e30f;
  for (int dh = -1; dh <= 1; ++dh) {
    int hh = h + dh; if (hh < 0 || hh >= HH) continue;
    for (int dw = -1; dw <= 1; ++dw) {
      int ww = w + dw; if (ww < 0 || ww >= WW) continue;
      m = fmaxf(m, base[hh * WW + ww]);
    }
  }
  qpT[((size_t)(b * NPOS + n)) * C8 + ch] = f2bf(m);
}

// ------------------------------------- flash pass 1: per-row max & sumexp
// Block = 8 waves = 128 queries.  Double-buffered 32-row K slab staged in LDS
// (next slab loaded to registers one step ahead); one barrier per step.
__global__ void __launch_bounds__(256) stats_kernel(
    const u16* __restrict__ qpT, const u16* __restrict__ kpT,
    float* __restrict__ mx, float* __restrict__ invl) {
  __shared__ u16 kTile[2][32 * C8];   // 8 KB
  int b = blockIdx.y;
  int w = threadIdx.x >> 5, l = threadIdx.x & 31;
  int lh = l >> 4, ll = l & 15;
  int m0 = (blockIdx.x * 8 + w) * 16;
  const u16* arow = qpT + ((size_t)b * NPOS + m0 + ll) * C8;
  v16bf a0 = ldfrag(arow, 0, lh);
  v16bf a1 = ldfrag(arow, 32, lh);
  float lm[8], ls[8];
#pragma unroll
  for (int i = 0; i < 8; ++i) { lm[i] = -1e30f; ls[i] = 0.f; }
  const u16* kbase = kpT + (size_t)b * NPOS * C8;
  uint4 stage = *reinterpret_cast<const uint4*>(kbase + threadIdx.x * 8);
  for (int n0 = 0; n0 < NPOS; n0 += 32) {
    int par = (n0 >> 5) & 1;
    *reinterpret_cast<uint4*>(&kTile[par][threadIdx.x * 8]) = stage;
    if (n0 + 32 < NPOS)
      stage = *reinterpret_cast<const uint4*>(kbase + (size_t)(n0 + 32) * C8 + threadIdx.x * 8);
    __syncthreads();
#pragma unroll
    for (int sub = 0; sub < 2; ++sub) {
      const u16* brow = &kTile[par][(sub * 16 + ll) * C8];
      v16bf b0 = ldfrag(brow, 0, lh);
      v16bf b1 = ldfrag(brow, 32, lh);
      v8f s = {};
      s = WMMA_BF16(a0, b0, s);
      s = WMMA_BF16(a1, b1, s);
#pragma unroll
      for (int i = 0; i < 8; ++i) {
        float v  = s[i];
        float nm = fmaxf(lm[i], v);
        ls[i] = ls[i] * __expf(lm[i] - nm) + __expf(v - nm);
        lm[i] = nm;
      }
    }
  }
  // combine the 16 lanes that share each row (rows 0-7 in lanes 0-15, 8-15 in 16-31)
  for (int mask = 1; mask <= 8; mask <<= 1) {
#pragma unroll
    for (int i = 0; i < 8; ++i) {
      float om = __shfl_xor(lm[i], mask, 32);
      float os = __shfl_xor(ls[i], mask, 32);
      float nm = fmaxf(lm[i], om);
      ls[i] = ls[i] * __expf(lm[i] - nm) + os * __expf(om - nm);
      lm[i] = nm;
    }
  }
  if (ll == 0) {
#pragma unroll
    for (int i = 0; i < 8; ++i) {
      int row = m0 + 8 * lh + i;
      mx[b * NPOS + row]   = lm[i];
      invl[b * NPOS + row] = 1.0f / ls[i];
    }
  }
}

// --------------- flash pass 2: recompute P tile, out = gamma*(V P^T) + x
// Block = 64 queries x all 512 channels.  Per 32-n step: each of the 8 waves
// computes one 16x16 score tile (4 m-tiles x 2 n-subtiles = 8 tasks), writes
// probabilities to LDS, then accumulates 4 c-tiles x 4 m-tiles (16 WMMAs).
// K slab + P tile double-buffered -> 2 barriers per step; next K slab is
// register-staged a full step ahead; V rows prefetched one step ahead.
__global__ void __launch_bounds__(256) attn_out_kernel(
    const u16* __restrict__ qpT, const u16* __restrict__ kpT,
    const u16* __restrict__ vbuf, const float* __restrict__ mx,
    const float* __restrict__ invl, const float* __restrict__ x,
    const float* __restrict__ gamma, float* __restrict__ out) {
  __shared__ u16 kTile[2][32 * C8];   // 8 KB
  __shared__ u16 pT[2][64 * 32];      // 8 KB, [m][n] probability tiles
  int b  = blockIdx.y;
  int m0 = blockIdx.x * 64;
  int w = threadIdx.x >> 5, l = threadIdx.x & 31;
  int lh = l >> 4, ll = l & 15;
  int mtj = w >> 1, sub = w & 1;   // this wave's score-tile task

  const u16* arow = qpT + ((size_t)b * NPOS + m0 + mtj * 16 + ll) * C8;
  v16bf aq0 = ldfrag(arow, 0, lh);
  v16bf aq1 = ldfrag(arow, 32, lh);
  float rmx[8], rin[8];
#pragma unroll
  for (int i = 0; i < 8; ++i) {
    rmx[i] = mx[b * NPOS + m0 + mtj * 16 + 8 * lh + i];
    rin[i] = invl[b * NPOS + m0 + mtj * 16 + 8 * lh + i];
  }
  const u16* kbase = kpT + (size_t)b * NPOS * C8;
  const u16* vr[4];
#pragma unroll
  for (int j = 0; j < 4; ++j)
    vr[j] = vbuf + ((size_t)b * CH + (w + 8 * j) * 16 + ll) * NPOS;
  v8f acc[4][4];
#pragma unroll
  for (int j = 0; j < 4; ++j)
#pragma unroll
    for (int jm = 0; jm < 4; ++jm) acc[j][jm] = (v8f){};

  uint4 stage = *reinterpret_cast<const uint4*>(kbase + threadIdx.x * 8);
  for (int n0 = 0; n0 < NPOS; n0 += 32) {
    int par = (n0 >> 5) & 1;
    *reinterpret_cast<uint4*>(&kTile[par][threadIdx.x * 8]) = stage;
    if (n0 + 32 < NPOS)
      stage = *reinterpret_cast<const uint4*>(kbase + (size_t)(n0 + 32) * C8 + threadIdx.x * 8);
    // prefetch next step's V rows while we compute this step
#pragma unroll
    for (int j = 0; j < 4; ++j)
      __builtin_prefetch((const void*)(vr[j] + n0 + 32), 0, 3);
    __syncthreads();
    // score tile: S[mtj*16.., n0+sub*16..] over K=64 channels
    {
      const u16* brow = &kTile[par][(sub * 16 + ll) * C8];
      v16bf b0 = ldfrag(brow, 0, lh);
      v16bf b1 = ldfrag(brow, 32, lh);
      v8f s = {};
      s = WMMA_BF16(aq0, b0, s);
      s = WMMA_BF16(aq1, b1, s);
#pragma unroll
      for (int i = 0; i < 8; ++i) {
        float p = __expf(s[i] - rmx[i]) * rin[i];
        pT[par][(mtj * 16 + 8 * lh + i) * 32 + sub * 16 + ll] = f2bf(p);
      }
    }
    __syncthreads();
    // hoisted B-fragments of P (4 LDS fragment builds instead of 16)
    v16bf bp[4];
#pragma unroll
    for (int jm = 0; jm < 4; ++jm)
      bp[jm] = ldfrag(&pT[par][(jm * 16 + ll) * 32], 0, lh);
    // output accumulation: 4 c-tiles x 4 m-tiles
#pragma unroll
    for (int j = 0; j < 4; ++j) {
      v16bf av = ldfrag(vr[j], n0, lh);
#pragma unroll
      for (int jm = 0; jm < 4; ++jm)
        acc[j][jm] = WMMA_BF16(av, bp[jm], acc[j][jm]);
    }
  }
  float g = gamma[0];
#pragma unroll
  for (int j = 0; j < 4; ++j) {
#pragma unroll
    for (int jm = 0; jm < 4; ++jm) {
      int col = m0 + jm * 16 + ll;
#pragma unroll
      for (int i = 0; i < 8; ++i) {
        int c = (w + 8 * j) * 16 + 8 * lh + i;
        size_t idx = ((size_t)(b * CH + c)) * NPOS + col;
        out[idx] = g * acc[j][jm][i] + x[idx];
      }
    }
  }
}

extern "C" void kernel_launch(void* const* d_in, const int* in_sizes, int n_in,
                              void* d_out, int out_size, void* d_ws, size_t ws_size,
                              hipStream_t stream) {
  (void)in_sizes; (void)n_in; (void)out_size; (void)ws_size;
  const float* x     = (const float*)d_in[0];
  const float* Wq    = (const float*)d_in[1];
  const float* bq    = (const float*)d_in[2];
  const float* Wk    = (const float*)d_in[3];
  const float* bk    = (const float*)d_in[4];
  const float* Wv    = (const float*)d_in[5];
  const float* bv    = (const float*)d_in[6];
  const float* gamma = (const float*)d_in[7];
  float* out = (float*)d_out;

  char* ws = (char*)d_ws;
  size_t off = 0;
  auto alloc = [&](size_t bytes) {
    void* p = ws + off;
    off = (off + bytes + 255) & ~(size_t)255;
    return p;
  };
  u16*   xT   = (u16*)  alloc((size_t)BATCH * NPOS * CH * 2);   // x^T bf16
  u16*   WqB  = (u16*)  alloc((size_t)C8 * CH * 2);
  u16*   WkB  = (u16*)  alloc((size_t)C8 * CH * 2);
  u16*   WvB  = (u16*)  alloc((size_t)CH * CH * 2);
  float* qc   = (float*)alloc((size_t)BATCH * C8 * NPOS * 4);   // pre-pool q
  float* kc   = (float*)alloc((size_t)BATCH * C8 * NPOS * 4);   // pre-pool k
  u16*   qpT  = (u16*)  alloc((size_t)BATCH * NPOS * C8 * 2);   // pooled q^T bf16
  u16*   kpT  = (u16*)  alloc((size_t)BATCH * NPOS * C8 * 2);   // pooled k^T bf16
  u16*   vbuf = (u16*)  alloc((size_t)BATCH * CH * NPOS * 2);   // v bf16 [c][n]
  float* mxb  = (float*)alloc((size_t)BATCH * NPOS * 4);        // row max
  float* invl = (float*)alloc((size_t)BATCH * NPOS * 4);        // 1/row sumexp

  xpose_kernel<<<dim3(NPOS / 32, CH / 32, BATCH), dim3(32, 8), 0, stream>>>(x, xT);
  cvt_kernel<<<(C8 * CH + 255) / 256, 256, 0, stream>>>(Wq, WqB, C8 * CH);
  cvt_kernel<<<(C8 * CH + 255) / 256, 256, 0, stream>>>(Wk, WkB, C8 * CH);
  cvt_kernel<<<(CH * CH + 255) / 256, 256, 0, stream>>>(Wv, WvB, CH * CH);

  gemm_conv_f32<<<dim3((C8 / 16) * NT / 8, BATCH), 256, 0, stream>>>(WqB, xT, bq, qc, C8);
  gemm_conv_f32<<<dim3((C8 / 16) * NT / 8, BATCH), 256, 0, stream>>>(WkB, xT, bk, kc, C8);
  pool_t_kernel<<<(BATCH * C8 * NPOS) / 256, 256, 0, stream>>>(qc, qpT);
  pool_t_kernel<<<(BATCH * C8 * NPOS) / 256, 256, 0, stream>>>(kc, kpT);
  gemm_conv_bf16<<<dim3((CH / 16) * NT / 8, BATCH), 256, 0, stream>>>(WvB, xT, bv, vbuf, CH);

  stats_kernel<<<dim3(NPOS / 128, BATCH), 256, 0, stream>>>(qpT, kpT, mxb, invl);
  attn_out_kernel<<<dim3(NPOS / 64, BATCH), 256, 0, stream>>>(qpT, kpT, vbuf, mxb, invl, x, gamma, out);
}
